// Experts_20349555049105
// MI455X (gfx1250) — matrix operand order repaired and checked by reading, not compiled, not run on hardware
//
#include <hip/hip_runtime.h>
#include <stdint.h>

// Problem constants (from reference)
#define N_TOK 4096   // B*S
#define DIM   2048   // D
#define FDIM  8192   // F
#define NEXP  8
#define RANK  8
#define TOPK  2
#define SCALING 2.0f // lora_alpha / r = 16/8

typedef __attribute__((ext_vector_type(16))) __bf16 v16bf;
typedef __attribute__((ext_vector_type(8)))  float  v8f;
typedef __attribute__((ext_vector_type(4)))  int    v4i_t;

union FragBF { v16bf v; uint4 u[2]; };

// ---------- bf16 helpers (raw-bit storage; RNE convert) ----------
__device__ inline unsigned short f2bf(float f) {
  unsigned int u = __float_as_uint(f);
  u += 0x7fffu + ((u >> 16) & 1u);
  return (unsigned short)(u >> 16);
}
__device__ inline float bf2f(unsigned short h) {
  return __uint_as_float(((unsigned int)h) << 16);
}

// gelu_new (tanh approximation), tanh via exp to stay on fast HW path
__device__ inline float gelu_new(float x) {
  float u = 0.7978845608028654f * (x + 0.044715f * x * x * x);
  float e = __expf(2.0f * u);
  float t = 1.0f - 2.0f / (e + 1.0f);   // tanh(u), safe at +-inf
  return 0.5f * x * (1.0f + t);
}

// ---------- async global->LDS (gfx1250) with compile-safe fallback ----------
#if defined(__HIP_DEVICE_COMPILE__) && __has_builtin(__builtin_amdgcn_global_load_async_to_lds_b128)
#define USE_ASYNC_LDS 1
typedef __attribute__((address_space(1))) v4i_t* gv4i_p;  // global int4*
typedef __attribute__((address_space(3))) v4i_t* lv4i_p;  // LDS int4*
#else
#define USE_ASYNC_LDS 0
#endif

__device__ inline void cp_b128(const void* g, void* l) {
#if USE_ASYNC_LDS
  __builtin_amdgcn_global_load_async_to_lds_b128((gv4i_p)g, (lv4i_p)l, 0, 0);
#else
  *(uint4*)l = *(const uint4*)g;
#endif
}

// Wait until at most N async ops remain outstanding (deferred wait enables
// DMA(tile i+1) to overlap WMMA(tile i); async loads retire in order).
template <int N>
__device__ inline void cp_wait_n() {
#if USE_ASYNC_LDS
#if __has_builtin(__builtin_amdgcn_s_wait_asynccnt)
  __builtin_amdgcn_s_wait_asynccnt(N);
#else
  asm volatile("s_wait_asynccnt %0" :: "i"(N) : "memory");
#endif
#endif
}

// ---------- block reduction (wave32 shuffle + LDS across 8 waves) ----------
__device__ inline float blockReduce256(float v, float* s8) {
  const int tid = threadIdx.x;
#pragma unroll
  for (int off = 16; off > 0; off >>= 1) v += __shfl_down(v, off, 32);
  __syncthreads();                 // protect s8 reuse across calls
  if ((tid & 31) == 0) s8[tid >> 5] = v;
  __syncthreads();
  return s8[0] + s8[1] + s8[2] + s8[3] + s8[4] + s8[5] + s8[6] + s8[7];
}

// ---------- fp32 -> bf16 conversion ----------
__global__ void cvt_f32_to_bf16(const float* __restrict__ in,
                                unsigned short* __restrict__ out, long long n) {
  long long i = ((long long)blockIdx.x * blockDim.x + threadIdx.x) * 4;
  if (i + 3 < n) {
    float4 v = *(const float4*)(in + i);
    ushort4 o;
    o.x = f2bf(v.x); o.y = f2bf(v.y); o.z = f2bf(v.z); o.w = f2bf(v.w);
    *(ushort4*)(out + i) = o;
  }
}

// ---------- router (softmax top-2) + z1 = x . A1[e]^T ----------
__global__ __launch_bounds__(256) void router_z1_kernel(
    const float* __restrict__ x,   // [N,D]
    const float* __restrict__ Wr,  // [E,D]
    const float* __restrict__ br,  // [E]
    const float* __restrict__ A1,  // [E,R,D]
    float* __restrict__ wtop,      // [N,2]
    int* __restrict__ idxo,        // [N,2]
    float* __restrict__ z1)        // [N,2,R]
{
  __shared__ float s8[8];
  __shared__ float slog[NEXP];
  __shared__ int   sidx[TOPK];
  const int n = blockIdx.x, tid = threadIdx.x;
  const float* xr = x + (size_t)n * DIM;
  float xv[8];
#pragma unroll
  for (int j = 0; j < 8; ++j) xv[j] = xr[tid * 8 + j];

  for (int e = 0; e < NEXP; ++e) {
    const float* w = Wr + (size_t)e * DIM + tid * 8;
    float p = 0.f;
#pragma unroll
    for (int j = 0; j < 8; ++j) p += xv[j] * w[j];
    float tot = blockReduce256(p, s8);
    if (tid == 0) slog[e] = tot + br[e];
  }
  __syncthreads();
  if (tid == 0) {
    float mx = slog[0];
    for (int e = 1; e < NEXP; ++e) mx = fmaxf(mx, slog[e]);
    float pe[NEXP]; float sum = 0.f;
    for (int e = 0; e < NEXP; ++e) { pe[e] = __expf(slog[e] - mx); sum += pe[e]; }
    int i0 = 0; float v0 = pe[0];
    for (int e = 1; e < NEXP; ++e) if (pe[e] > v0) { v0 = pe[e]; i0 = e; }
    int i1 = -1; float v1 = -1.f;
    for (int e = 0; e < NEXP; ++e) if (e != i0 && pe[e] > v1) { v1 = pe[e]; i1 = e; }
    sidx[0] = i0; sidx[1] = i1;
    wtop[n * 2 + 0] = v0 / sum; wtop[n * 2 + 1] = v1 / sum;
    idxo[n * 2 + 0] = i0;       idxo[n * 2 + 1] = i1;
  }
  __syncthreads();
  for (int k = 0; k < TOPK; ++k) {
    const int e = sidx[k];
    for (int r = 0; r < RANK; ++r) {
      const float* a = A1 + ((size_t)(e * RANK + r)) * DIM + tid * 8;
      float p = 0.f;
#pragma unroll
      for (int j = 0; j < 8; ++j) p += xv[j] * a[j];
      float tot = blockReduce256(p, s8);
      if (tid == 0) z1[(n * 2 + k) * RANK + r] = tot;
    }
  }
}

// ---------- z2 = h_k . A2[e]^T ----------
__global__ __launch_bounds__(256) void z2_kernel(
    const unsigned short* __restrict__ h, // [2,N,F] bf16
    const float* __restrict__ A2,         // [E,R,F]
    const int* __restrict__ idxo,         // [N,2]
    float* __restrict__ z2)               // [N,2,R]
{
  __shared__ float s8[8];
  const int n = blockIdx.x, k = blockIdx.y, tid = threadIdx.x;
  const int e = idxo[n * 2 + k];
  const unsigned short* hr = h + ((size_t)k * N_TOK + n) * FDIM;
  float acc[RANK];
#pragma unroll
  for (int r = 0; r < RANK; ++r) acc[r] = 0.f;
  for (int j = 0; j < FDIM / 256; ++j) {
    const int f = tid + j * 256;
    const float hv = bf2f(hr[f]);
#pragma unroll
    for (int r = 0; r < RANK; ++r)
      acc[r] += hv * A2[((size_t)(e * RANK + r)) * FDIM + f];
  }
  for (int r = 0; r < RANK; ++r) {
    float tot = blockReduce256(acc[r], s8);
    if (tid == 0) z2[(n * 2 + k) * RANK + r] = tot;
  }
}

// ---------- WMMA bf16 GEMM (NT): C[m,n] = sum_k A[m,k]*Bt[n,k], fused epilogues ----
// Double-buffered LDS: async DMA of tile i+1 overlaps WMMA of tile i.
// EPI==1: h_k = gelu(acc + b1 + SCALING*z1_k.B1[e_k,col,:]) -> bf16 x2
// EPI==2: out (=/+=) w_k*(acc + b2 + SCALING*z2_k.B2[e_k,col,:])
template <int EPI>
__global__ __launch_bounds__(256) void gemm_wmma_bf16(
    const unsigned short* __restrict__ A,   // [M,Kd] bf16
    const unsigned short* __restrict__ Bt,  // [Nd,Kd] bf16
    int M, int Nd, int Kd,
    const float* __restrict__ bias,         // [Nd]
    const int* __restrict__ idx,            // [M,2]
    unsigned short* __restrict__ hOut,      // EPI1: [2,M,Nd]
    const float* __restrict__ z1,           // EPI1: [M,2,8]
    const float* __restrict__ B1,           // EPI1: [E,Nd,8]
    float* __restrict__ out,                // EPI2: [M,Nd]
    const float* __restrict__ z2,           // EPI2: [M,2,8]
    const float* __restrict__ B2,           // EPI2: [E,Nd,8]
    const float* __restrict__ wtop,         // EPI2: [M,2]
    int kslot)
{
  __shared__ alignas(16) unsigned short As[2][128][40];  // 32 used + 8 pad, x2 buffers
  __shared__ alignas(16) unsigned short Bs[2][128][40];

  const int tid  = threadIdx.x;
  const int lane = tid & 31;
  const int wv   = tid >> 5;     // 8 waves
  const int wm   = wv >> 2;      // 0..1 -> 64 rows each
  const int wn   = wv & 3;       // 0..3 -> 32 cols each
  const int rowBlk = blockIdx.y * 128;
  const int colBlk = blockIdx.x * 128;
  const int r16   = lane & 15;
  const int halfk = lane >> 4;

  // Per-thread cooperative-load coordinates (2 x 16B chunks per matrix)
  const int c0 = tid, c1 = tid + 256;
  const int r0 = c0 >> 2, co0 = (c0 & 3) * 8;
  const int r1 = c1 >> 2, co1 = (c1 & 3) * 8;
  const unsigned short* gA0 = A  + (size_t)(rowBlk + r0) * Kd + co0;
  const unsigned short* gA1 = A  + (size_t)(rowBlk + r1) * Kd + co1;
  const unsigned short* gB0 = Bt + (size_t)(colBlk + r0) * Kd + co0;
  const unsigned short* gB1 = Bt + (size_t)(colBlk + r1) * Kd + co1;

  v8f acc[4][2];
  const v8f vzero = {0.f, 0.f, 0.f, 0.f, 0.f, 0.f, 0.f, 0.f};
#pragma unroll
  for (int mt = 0; mt < 4; ++mt)
#pragma unroll
    for (int nt = 0; nt < 2; ++nt) acc[mt][nt] = vzero;

  // Prologue: stage tile kk=0 into buffer 0 (4 async ops per thread)
  cp_b128(gA0, &As[0][r0][co0]);
  cp_b128(gA1, &As[0][r1][co1]);
  cp_b128(gB0, &Bs[0][r0][co0]);
  cp_b128(gB1, &Bs[0][r1][co1]);

  int buf = 0;
  for (int kk = 0; kk < Kd; kk += 32) {
    if (kk + 32 < Kd) {
      // Issue next tile into the other buffer (safe: it was last read in the
      // previous iteration, behind that iteration's trailing barrier), then
      // wait only for the current tile's 4 in-order async ops.
      const int nb = buf ^ 1;
      const int kn = kk + 32;
      cp_b128(gA0 + kn, &As[nb][r0][co0]);
      cp_b128(gA1 + kn, &As[nb][r1][co1]);
      cp_b128(gB0 + kn, &Bs[nb][r0][co0]);
      cp_b128(gB1 + kn, &Bs[nb][r1][co1]);
      cp_wait_n<4>();
    } else {
      cp_wait_n<0>();
    }
    __syncthreads();

    // B fragments: lane = column, half-wave selects K 0..15 / 16..31
    FragBF bfr[2];
#pragma unroll
    for (int nt = 0; nt < 2; ++nt) {
      const int rb = wn * 32 + nt * 16 + r16;
      bfr[nt].u[0] = *(const uint4*)&Bs[buf][rb][halfk * 16];
      bfr[nt].u[1] = *(const uint4*)&Bs[buf][rb][halfk * 16 + 8];
    }
    // A fragments: lane = row, K = h*16..h*16+15
#pragma unroll
    for (int mt = 0; mt < 4; ++mt) {
      FragBF afr;
      const int ra = wm * 64 + mt * 16 + r16;
      afr.u[0] = *(const uint4*)&As[buf][ra][halfk * 16];
      afr.u[1] = *(const uint4*)&As[buf][ra][halfk * 16 + 8];
#pragma unroll
      for (int nt = 0; nt < 2; ++nt)
        acc[mt][nt] = __builtin_amdgcn_wmma_f32_16x16x32_bf16(
            false, afr.v, false, bfr[nt].v, (short)0, acc[mt][nt], false, false);
    }
    __syncthreads();
    buf ^= 1;
  }

  // Epilogue. C/D layout: lane<16 -> N=lane, M=vgpr j; lane>=16 -> N=lane-16, M=8+j
#pragma unroll
  for (int mt = 0; mt < 4; ++mt) {
#pragma unroll
    for (int nt = 0; nt < 2; ++nt) {
      const int col  = colBlk + wn * 32 + nt * 16 + r16;
      const int row0 = rowBlk + wm * 64 + mt * 16 + halfk * 8;
      const float bcol = bias[col];
#pragma unroll
      for (int j = 0; j < 8; ++j) {
        const int row = row0 + j;
        const float base = acc[mt][nt][j] + bcol;
        if (EPI == 1) {
#pragma unroll
          for (int k = 0; k < TOPK; ++k) {
            const int e = idx[row * TOPK + k];
            const float* zp = z1 + (size_t)(row * TOPK + k) * RANK;
            const float* bp = B1 + ((size_t)e * Nd + col) * RANK;
            float lo = 0.f;
#pragma unroll
            for (int r = 0; r < RANK; ++r) lo += zp[r] * bp[r];
            hOut[((size_t)k * M + row) * Nd + col] = f2bf(gelu_new(base + SCALING * lo));
          }
        } else {
          const int e = idx[row * TOPK + kslot];
          const float w = wtop[row * TOPK + kslot];
          const float* zp = z2 + (size_t)(row * TOPK + kslot) * RANK;
          const float* bp = B2 + ((size_t)e * Nd + col) * RANK;
          float lo = 0.f;
#pragma unroll
          for (int r = 0; r < RANK; ++r) lo += zp[r] * bp[r];
          const float y = base + SCALING * lo;
          float* op = out + (size_t)row * Nd + col;
          if (kslot == 0) *op = w * y;
          else            *op += w * y;
        }
      }
    }
  }
}

extern "C" void kernel_launch(void* const* d_in, const int* in_sizes, int n_in,
                              void* d_out, int out_size, void* d_ws, size_t ws_size,
                              hipStream_t stream) {
  (void)in_sizes; (void)n_in; (void)out_size; (void)ws_size;
  const float* x  = (const float*)d_in[0];
  const float* Wr = (const float*)d_in[1];
  const float* br = (const float*)d_in[2];
  const float* W1 = (const float*)d_in[3];
  const float* b1 = (const float*)d_in[4];
  const float* W2 = (const float*)d_in[5];
  const float* b2 = (const float*)d_in[6];
  const float* A1 = (const float*)d_in[7];
  const float* B1 = (const float*)d_in[8];
  const float* A2 = (const float*)d_in[9];
  const float* B2 = (const float*)d_in[10];
  float* out = (float*)d_out;

  // Workspace carving (~219 MB)
  char* ws = (char*)d_ws;
  unsigned short* xb  = (unsigned short*)ws; ws += (size_t)N_TOK * DIM * 2;
  unsigned short* W1b = (unsigned short*)ws; ws += (size_t)FDIM * DIM * 2;
  unsigned short* W2b = (unsigned short*)ws; ws += (size_t)DIM * FDIM * 2;
  unsigned short* h   = (unsigned short*)ws; ws += (size_t)2 * N_TOK * FDIM * 2;
  float* z1   = (float*)ws; ws += (size_t)N_TOK * TOPK * RANK * 4;
  float* z2   = (float*)ws; ws += (size_t)N_TOK * TOPK * RANK * 4;
  float* wtop = (float*)ws; ws += (size_t)N_TOK * TOPK * 4;
  int*   idx  = (int*)ws;

  // 1) fp32 -> bf16 operands
  {
    long long nx = (long long)N_TOK * DIM;
    long long nw = (long long)FDIM * DIM;
    cvt_f32_to_bf16<<<(unsigned)((nx + 1023) / 1024), 256, 0, stream>>>(x,  xb,  nx);
    cvt_f32_to_bf16<<<(unsigned)((nw + 1023) / 1024), 256, 0, stream>>>(W1, W1b, nw);
    cvt_f32_to_bf16<<<(unsigned)((nw + 1023) / 1024), 256, 0, stream>>>(W2, W2b, nw);
  }

  // 2) router + z1
  router_z1_kernel<<<N_TOK, 256, 0, stream>>>(x, Wr, br, A1, wtop, idx, z1);

  // 3) fc1 GEMM + fused lora1 + gelu -> h[2][N][F] bf16
  {
    dim3 g(FDIM / 128, N_TOK / 128);
    gemm_wmma_bf16<1><<<g, 256, 0, stream>>>(
        xb, W1b, N_TOK, FDIM, DIM, b1, idx,
        h, z1, B1, nullptr, nullptr, nullptr, nullptr, 0);
  }

  // 4) z2 reductions for both top-k slots
  {
    dim3 g(N_TOK, TOPK);
    z2_kernel<<<g, 256, 0, stream>>>(h, A2, idx, z2);
  }

  // 5) fc2 GEMM x2 + fused lora2 + gated accumulate into out
  {
    dim3 g(DIM / 128, N_TOK / 128);
    for (int kslot = 0; kslot < TOPK; ++kslot) {
      gemm_wmma_bf16<2><<<g, 256, 0, stream>>>(
          h + (size_t)kslot * N_TOK * FDIM, W2b, N_TOK, DIM, FDIM, b2, idx,
          nullptr, nullptr, nullptr, out, z2, B2, wtop, kslot);
    }
  }
}
